// Attention_82712480186499
// MI455X (gfx1250) — matrix operand order repaired and checked by reading, not compiled, not run on hardware
//
#include <hip/hip_runtime.h>

typedef __attribute__((ext_vector_type(16))) _Float16     v16h;
typedef __attribute__((ext_vector_type(8)))  float        v8f;
typedef __attribute__((ext_vector_type(4)))  unsigned int v4u;
typedef __attribute__((ext_vector_type(8)))  int          v8i;
typedef __attribute__((ext_vector_type(4)))  int          v4i;

#define B_  4
#define H_  4
#define S_  2048
#define D_  64
#define QT  32      // q rows per workgroup (score tile rows resident in LDS)
#define KC  64      // K/V staging chunk (rows), double buffered f32 via TDM
#define NBH 16      // num_buckets/2 (bidirectional T5)

// LDS layout (bytes)
#define OFF_SS 0                                   // [QT][S_] f32   = 262144
#define OFF_QS (QT * S_ * 4)                       // [QT][D_] f16   = 4096
#define OFF_B0 (OFF_QS + QT * D_ * 2)              // [KC][D_] f32 x2 buffers
#define BUFB   (KC * D_ * 4)                       // 16384 per buffer
#define SMEM_BYTES (OFF_B0 + 2 * BUFB)             // 299008 < 320KB

// ---------------------------------------------------------------------------
// Precompute bias_tab[h][delta + S-1] = bias_table[bucket(delta)][h]
// ---------------------------------------------------------------------------
__global__ void bias_precompute_kernel(const float* __restrict__ bias_table,
                                       float* __restrict__ bias_tab) {
  int idx = blockIdx.x * blockDim.x + threadIdx.x;
  const int n = H_ * (2 * S_ - 1);
  if (idx >= n) return;
  int h = idx / (2 * S_ - 1);
  int d = idx % (2 * S_ - 1) - (S_ - 1);        // d = k - q
  int bucket = (d > 0) ? NBH : 0;
  int rp = d < 0 ? -d : d;
  const int max_exact = NBH / 2;                 // 8
  int bsm;
  if (rp < max_exact) {
    bsm = rp;
  } else {
    float lg = logf((float)rp / (float)max_exact) /
               logf(128.0f / (float)max_exact) * (float)(NBH - max_exact);
    int lb = max_exact + (int)lg;                // trunc toward zero
    bsm = lb < (NBH - 1) ? lb : (NBH - 1);
  }
  bias_tab[idx] = bias_table[(bucket + bsm) * H_ + h];
}

// ---------------------------------------------------------------------------
// TDM: 1-row tile DMA of nElems f32 from global -> LDS. D# per ISA ch.8.
// Issued by one wave; tracked with TENSORcnt.
// ---------------------------------------------------------------------------
__device__ __forceinline__ void tdm_load_1d(unsigned lds_off, const float* gsrc,
                                            int nElems) {
  unsigned long long ga = (unsigned long long)(size_t)gsrc;
  v4u g0;
  g0.x = 1u;                                         // count=1, is_restore=0, no gather
  g0.y = lds_off;                                    // [63:32]  lds_addr (bytes)
  g0.z = (unsigned)ga;                               // [95:64]  global_addr lo
  g0.w = (unsigned)((ga >> 32) & 0x1FFFFFFu)         // [120:96] global_addr hi (57-bit)
       | (2u << 30);                                 // [127:126] type=2 ("image")
  v8i g1;
  g1[0] = 2 << 16;                                   // workgroup_mask=0, data_size=4B
  g1[1] = (nElems & 0xFFFF) << 16;                   // tensor_dim0[15:0]
  g1[2] = ((nElems >> 16) & 0xFFFF) | (1 << 16);     // tensor_dim0[31:16] | tensor_dim1=1
  g1[3] = (nElems & 0xFFFF) << 16;                   // tile_dim0
  g1[4] = 1;                                         // tile_dim1=1, tile_dim2=0
  g1[5] = nElems;                                    // tensor_dim0_stride lo
  g1[6] = 0;
  g1[7] = 0;
  v4i z4 = {0, 0, 0, 0};
  v8i z8 = {0, 0, 0, 0, 0, 0, 0, 0};
  __builtin_amdgcn_tensor_load_to_lds(g0, g1, z4, z4, z8, 0);
}

// ---------------------------------------------------------------------------
// Fused attention: one workgroup per (b, h, 32-row q-tile); 8 wave32s.
// ---------------------------------------------------------------------------
__global__ void __launch_bounds__(256)
attn_kernel(const float* __restrict__ qg, const float* __restrict__ kg,
            const float* __restrict__ vg, const int* __restrict__ mask,
            const float* __restrict__ bias_tab,
            float* __restrict__ outg, float* __restrict__ pg) {
  extern __shared__ char smem[];
  float*    Ss    = (float*)(smem + OFF_SS);         // [QT][S_]
  _Float16* Qs    = (_Float16*)(smem + OFF_QS);      // [QT][D_]
  float*    KbufB = (float*)(smem + OFF_B0);         // 2x [KC][D_] f32
  // Runtime scalar LDS byte offset of buffer 0 (low 32 bits = LDS offset).
  unsigned ldsOffB0 = (unsigned)(size_t)(smem + OFF_B0);

  const int tid  = threadIdx.x;
  const int wave = tid >> 5;
  const int lane = tid & 31;
  const int q0   = blockIdx.x * QT;
  const int h    = blockIdx.y;
  const int b    = blockIdx.z;
  const long headBase = (long)(b * H_ + h) * S_ * D_;

  // WMMA per-lane fragment coordinates (ISA 7.12.2)
  const int m     = lane & 15;          // A row
  const int kb8   = (lane >> 4) * 8;    // A K-split
  const int kb16  = (lane >> 4) * 16;   // unused; B also uses kb8
  const int n16   = lane & 15;          // B/C/D column
  const int rowHi = (lane >> 4) * 8;    // C/D: M = vgpr + 8*(lane>=16)

  const int nch = S_ / KC;              // 32 chunks

  // ---- Phase 0: Q tile -> f16 LDS; kick off K chunk 0 DMA ----
  if (tid == 0) tdm_load_1d(ldsOffB0, kg + headBase, KC * D_);
  for (int i = tid; i < QT * D_; i += 256) {
    int r = i >> 6, c = i & 63;
    Qs[i] = (_Float16)qg[headBase + (long)(q0 + r) * D_ + c];
  }
  __syncthreads();

  v16h aQ[2][2];                        // [rowblock][k-chunk of 32]
  #pragma unroll
  for (int rb = 0; rb < 2; ++rb) {
    #pragma unroll
    for (int kc2 = 0; kc2 < 2; ++kc2) {
      v16h a;
      int base = (rb * 16 + m) * D_ + kc2 * 32 + kb8;
      #pragma unroll
      for (int i = 0; i < 8; ++i) {
        a[i]     = Qs[base + i];
        a[8 + i] = Qs[base + 16 + i];
      }
      aQ[rb][kc2] = a;
    }
  }

  // ---- Phase 1: scores = (Q K^T + bias)/8, masked; TDM double-buffered K ----
  for (int c = 0; c < nch; ++c) {
    if (tid == 0) {
      if (c + 1 < nch) {
        tdm_load_1d(ldsOffB0 + (unsigned)(((c + 1) & 1) * BUFB),
                    kg + headBase + (long)(c + 1) * KC * D_, KC * D_);
        __builtin_amdgcn_s_wait_tensorcnt(1);   // chunk c ready (in-order)
      } else {
        __builtin_amdgcn_s_wait_tensorcnt(0);
      }
    }
    __syncthreads();
    const float* Ks = KbufB + (c & 1) * (KC * D_);
    const int kc = c * KC;

    // 2 rowblocks x 4 colblocks = 8 tiles; 1 per wave
    {
      int rb = wave & 1, cb = wave >> 1;
      v8f acc = {};
      #pragma unroll
      for (int kc2 = 0; kc2 < 2; ++kc2) {
        v16h bf;                        // B[k][n] = K[cb*16+n][kc2*32 + k]
        int kbase = (cb * 16 + n16) * D_ + kc2 * 32 + kb8;
        #pragma unroll
        for (int i = 0; i < 16; ++i) bf[i] = (_Float16)Ks[kbase + i + (i & 8)];
        acc = __builtin_amdgcn_wmma_f32_16x16x32_f16(
            false, aQ[rb][kc2], false, bf, (short)0, acc, false, false);
      }
      int kcol = kc + cb * 16 + n16;
      int mk   = mask[b * S_ + kcol];
      const float* bt = bias_tab + h * (2 * S_ - 1) + (kcol + S_ - 1);
      #pragma unroll
      for (int r = 0; r < 8; ++r) {
        int row  = rb * 16 + rowHi + r;
        int qrow = q0 + row;
        float s  = (acc[r] + bt[-qrow]) * 0.125f;   // 1/sqrt(64)
        if (mk == 0) s = -1e9f;
        Ss[row * S_ + kcol] = s;
      }
    }
    __syncthreads();
  }

  // Overlap: start V chunk 0 DMA while softmax runs (K buffers are free now)
  if (tid == 0) tdm_load_1d(ldsOffB0, vg + headBase, KC * D_);

  // ---- Phase 2: row softmax in LDS; write normalized p_attn ----
  for (int rr = 0; rr < 4; ++rr) {
    int row = wave * 4 + rr;
    float* Sr = Ss + row * S_;
    float mx = -3.4e38f;
    for (int i = lane; i < S_; i += 32) mx = fmaxf(mx, Sr[i]);
    #pragma unroll
    for (int off = 16; off > 0; off >>= 1) mx = fmaxf(mx, __shfl_xor(mx, off, 32));
    float sum = 0.f;
    for (int i = lane; i < S_; i += 32) {
      float e = __expf(Sr[i] - mx);
      Sr[i] = e;
      sum += e;
    }
    #pragma unroll
    for (int off = 16; off > 0; off >>= 1) sum += __shfl_xor(sum, off, 32);
    float inv = 1.0f / sum;
    long prow = ((long)(b * H_ + h) * S_ + q0 + row) * S_;
    for (int i = lane; i < S_; i += 32) {
      float p = Sr[i] * inv;
      Sr[i] = p;                        // keep normalized probs for P@V
      pg[prow + i] = p;                 // coalesced
    }
  }
  __syncthreads();

  // ---- Phase 3: out = P @ V; TDM double-buffered V ----
  const int rb  = wave & 1;
  const int cbv = wave >> 1;            // D/16 = 4 colblocks
  v8f oacc = {};
  for (int c = 0; c < nch; ++c) {
    if (tid == 0) {
      if (c + 1 < nch) {
        tdm_load_1d(ldsOffB0 + (unsigned)(((c + 1) & 1) * BUFB),
                    vg + headBase + (long)(c + 1) * KC * D_, KC * D_);
        __builtin_amdgcn_s_wait_tensorcnt(1);
      } else {
        __builtin_amdgcn_s_wait_tensorcnt(0);
      }
    }
    __syncthreads();
    const float* Vs = KbufB + (c & 1) * (KC * D_);
    const int vc = c * KC;
    #pragma unroll
    for (int kk = 0; kk < KC; kk += 32) {
      v16h a, bf;
      int abase = (rb * 16 + m) * S_ + vc + kk + kb8;
      #pragma unroll
      for (int i = 0; i < 8; ++i) {
        a[i]     = (_Float16)Ss[abase + i];
        a[8 + i] = (_Float16)Ss[abase + 16 + i];
      }
      int bbase = (kk + kb8) * D_ + cbv * 16 + n16;
      #pragma unroll
      for (int i = 0; i < 16; ++i) bf[i] = (_Float16)Vs[bbase + (i + (i & 8)) * D_];
      oacc = __builtin_amdgcn_wmma_f32_16x16x32_f16(
          false, a, false, bf, (short)0, oacc, false, false);
    }
    __syncthreads();
  }
  long obase = (long)(b * H_ + h) * S_ + q0;
  #pragma unroll
  for (int r = 0; r < 8; ++r) {
    int row = rb * 16 + rowHi + r;
    outg[(obase + row) * D_ + cbv * 16 + n16] = oacc[r];
  }
}

// ---------------------------------------------------------------------------
extern "C" void kernel_launch(void* const* d_in, const int* in_sizes, int n_in,
                              void* d_out, int out_size, void* d_ws, size_t ws_size,
                              hipStream_t stream) {
  const float* q          = (const float*)d_in[0];
  const float* k          = (const float*)d_in[1];
  const float* v          = (const float*)d_in[2];
  const int*   mask       = (const int*)d_in[3];
  const float* bias_table = (const float*)d_in[4];

  float* out      = (float*)d_out;                   // [B,H,S,D]
  float* p_attn   = out + (long)B_ * H_ * S_ * D_;   // [B,H,S,S]
  float* bias_tab = (float*)d_ws;                    // H*(2S-1) floats

  const int nbias = H_ * (2 * S_ - 1);
  bias_precompute_kernel<<<(nbias + 255) / 256, 256, 0, stream>>>(bias_table, bias_tab);

  dim3 grid(S_ / QT, H_, B_);
  attn_kernel<<<grid, 256, SMEM_BYTES, stream>>>(q, k, v, mask, bias_tab, out, p_attn);
}
